// KmeansModel_60455959658906
// MI455X (gfx1250) — matrix operand order repaired and checked by reading, not compiled, not run on hardware
//
#include <hip/hip_runtime.h>
#include <hip/hip_bf16.h>

typedef __attribute__((ext_vector_type(16))) __bf16 v16bf;
typedef __attribute__((ext_vector_type(8)))  float  v8f;

#define D_DIM     768
#define TILE_ROWS 64
#define KTILE     128
#define DCHUNK    32
#define LDA       (D_DIM + 8)   // 776 halfs: row stride 1552B -> conflict-free frag reads

__device__ __forceinline__ unsigned short f32_to_bf16_rne(float f) {
  unsigned u = __float_as_uint(f);
  unsigned r = u + 0x7FFFu + ((u >> 16) & 1u);
  return (unsigned short)(r >> 16);
}

// ---------------- kernel 1: centroid squared norms ----------------
__global__ void cnorm_kernel(const float* __restrict__ c, float* __restrict__ cn,
                             int K, int D) {
  int k = blockIdx.x * blockDim.x + threadIdx.x;
  if (k >= K) return;
  float s = 0.f;
  for (int d = 0; d < D; ++d) {
    float v = c[(size_t)d * K + k];       // coalesced over k
    s = fmaf(v, v, s);
  }
  cn[k] = s;
}

// -- kernel 2: one-time transpose + fp32 -> bf16 hi/lo split of centroids --
// input  cent [D][K] row-major fp32
// output chi/clo [Kpad][D] (column-major w.r.t. original), zero-padded cols
__launch_bounds__(256)
__global__ void convert_centroids_kernel(const float* __restrict__ cent,
                                         unsigned short* __restrict__ chi,
                                         unsigned short* __restrict__ clo,
                                         int K) {
  __shared__ unsigned short shi[64 * 65];
  __shared__ unsigned short slo[64 * 65];
  const int t  = threadIdx.x;
  const int k0 = blockIdx.x * 64;
  const int d0 = blockIdx.y * 64;
#pragma unroll
  for (int i = 0; i < 16; ++i) {
    int idx = t + i * 256;
    int dr = idx >> 6, kc = idx & 63;
    int k = k0 + kc;
    float v = (k < K) ? cent[(size_t)(d0 + dr) * K + k] : 0.f;  // coalesced over k
    unsigned short hb = f32_to_bf16_rne(v);
    float hf = __uint_as_float((unsigned)hb << 16);
    unsigned short lb = f32_to_bf16_rne(v - hf);
    shi[kc * 65 + dr] = hb;
    slo[kc * 65 + dr] = lb;
  }
  __syncthreads();
#pragma unroll
  for (int i = 0; i < 16; ++i) {
    int idx = t + i * 256;
    int kr = idx >> 6, dc = idx & 63;
    size_t o = (size_t)(k0 + kr) * D_DIM + (d0 + dc);           // coalesced over d
    chi[o] = shi[kr * 65 + dc];
    clo[o] = slo[kr * 65 + dc];
  }
}

// ---------------- kernel 3: fused GEMM + running argmin ----------------
// 8 waves: 2 row-halves x 4 col-groups; each wave owns 32 rows x 32 cols
// (2 A row-frags x 2 B col-frags -> 4 accumulators, 12 WMMAs per d-chunk).
// A (x tile, bf16 hi/lo) persistent in LDS; B frags loaded straight from
// global (pre-converted, fragment-ready layout). K loop is barrier-free.
__launch_bounds__(256)
__global__ void kmeans_argmin_kernel(const float* __restrict__ x,
                                     const unsigned short* __restrict__ chi,
                                     const unsigned short* __restrict__ clo,
                                     const float* __restrict__ cnorm,
                                     int* __restrict__ out,
                                     int K, int Kpad) {
  __shared__ unsigned short ah[TILE_ROWS * LDA];
  __shared__ unsigned short al[TILE_ROWS * LDA];
  __shared__ float redD[4][TILE_ROWS];
  __shared__ int   redI[4][TILE_ROWS];

  const int t    = threadIdx.x;
  const int wave = t >> 5;
  const int lane = t & 31;
  const int rw   = wave >> 2;            // row half 0..1 (32 rows each)
  const int cw   = wave & 3;             // col group 0..3 (32 cols each)
  const int m    = lane & 15;
  const int h    = lane >> 4;
  const int rowBase = blockIdx.x * TILE_ROWS;

  // ---- one-time A stage: 64 rows x 768 d, fp32 -> bf16 hi/lo ----
#pragma unroll
  for (int i = 0; i < (TILE_ROWS * D_DIM) / 256; ++i) {
    int idx = t + i * 256;
    int r = idx / D_DIM, dd = idx % D_DIM;
    float v = x[(size_t)(rowBase + r) * D_DIM + dd];
    unsigned short hb = f32_to_bf16_rne(v);
    float hf = __uint_as_float((unsigned)hb << 16);
    unsigned short lb = f32_to_bf16_rne(v - hf);
    ah[r * LDA + dd] = hb;
    al[r * LDA + dd] = lb;
  }
  __syncthreads();   // only barrier before the K loop

  float bestD[16];
  int   bestI[16];
#pragma unroll
  for (int i = 0; i < 16; ++i) { bestD[i] = __builtin_inff(); bestI[i] = 0x7FFFFFFF; }

  // per-lane A row pointers (frag0: rows rw*32+m, frag1: +16)
  const unsigned short* arow0h = &ah[(rw * 32 + m) * LDA];
  const unsigned short* arow0l = &al[(rw * 32 + m) * LDA];
  const unsigned short* arow1h = arow0h + 16 * LDA;
  const unsigned short* arow1l = arow0l + 16 * LDA;

  for (int k0 = 0; k0 < Kpad; k0 += KTILE) {
    const int kcA = k0 + cw * 32 + m;     // this lane's two columns
    const int kcB = kcA + 16;
    const unsigned short* bc0h = chi + (size_t)kcA * D_DIM + h * 8;
    const unsigned short* bc0l = clo + (size_t)kcA * D_DIM + h * 8;
    const unsigned short* bc1h = chi + (size_t)kcB * D_DIM + h * 8;
    const unsigned short* bc1l = clo + (size_t)kcB * D_DIM + h * 8;
    if (k0 + KTILE < Kpad) {              // prefetch next tile's columns
      __builtin_prefetch(chi + (size_t)(kcA + KTILE) * D_DIM, 0, 1);
      __builtin_prefetch(clo + (size_t)(kcA + KTILE) * D_DIM, 0, 1);
      __builtin_prefetch(chi + (size_t)(kcB + KTILE) * D_DIM, 0, 1);
      __builtin_prefetch(clo + (size_t)(kcB + KTILE) * D_DIM, 0, 1);
    }

    v8f acc00 = {}; v8f acc01 = {};       // rows frag0 x cols A/B
    v8f acc10 = {}; v8f acc11 = {};       // rows frag1 x cols A/B
#pragma unroll 2
    for (int d0 = 0; d0 < D_DIM; d0 += DCHUNK) {
      union fragA { v16bf v; unsigned u[8]; } fa0h, fa0l, fa1h, fa1l;
      union fragB { v16bf v; uint4 q[2]; } fb0h, fb0l, fb1h, fb1l;
      {
        const unsigned* p0h = (const unsigned*)(arow0h + d0 + h * 8);
        const unsigned* p0l = (const unsigned*)(arow0l + d0 + h * 8);
        const unsigned* p1h = (const unsigned*)(arow1h + d0 + h * 8);
        const unsigned* p1l = (const unsigned*)(arow1l + d0 + h * 8);
#pragma unroll
        for (int j = 0; j < 4; ++j) {
          fa0h.u[j] = p0h[j]; fa0h.u[4 + j] = p0h[8 + j];
          fa0l.u[j] = p0l[j]; fa0l.u[4 + j] = p0l[8 + j];
          fa1h.u[j] = p1h[j]; fa1h.u[4 + j] = p1h[8 + j];
          fa1l.u[j] = p1l[j]; fa1l.u[4 + j] = p1l[8 + j];
        }
      }
      {
        const uint4* q0h = (const uint4*)(bc0h + d0);
        const uint4* q0l = (const uint4*)(bc0l + d0);
        const uint4* q1h = (const uint4*)(bc1h + d0);
        const uint4* q1l = (const uint4*)(bc1l + d0);
        fb0h.q[0] = q0h[0]; fb0h.q[1] = q0h[2];
        fb0l.q[0] = q0l[0]; fb0l.q[1] = q0l[2];
        fb1h.q[0] = q1h[0]; fb1h.q[1] = q1h[2];
        fb1l.q[0] = q1l[0]; fb1l.q[1] = q1l[2];
      }
      // 3xBF16 split accumulation (drop lo*lo), 4 output tiles
      acc00 = __builtin_amdgcn_wmma_f32_16x16x32_bf16(false, fa0h.v, false, fb0h.v, (short)0, acc00, false, false);
      acc00 = __builtin_amdgcn_wmma_f32_16x16x32_bf16(false, fa0h.v, false, fb0l.v, (short)0, acc00, false, false);
      acc00 = __builtin_amdgcn_wmma_f32_16x16x32_bf16(false, fa0l.v, false, fb0h.v, (short)0, acc00, false, false);
      acc01 = __builtin_amdgcn_wmma_f32_16x16x32_bf16(false, fa0h.v, false, fb1h.v, (short)0, acc01, false, false);
      acc01 = __builtin_amdgcn_wmma_f32_16x16x32_bf16(false, fa0h.v, false, fb1l.v, (short)0, acc01, false, false);
      acc01 = __builtin_amdgcn_wmma_f32_16x16x32_bf16(false, fa0l.v, false, fb1h.v, (short)0, acc01, false, false);
      acc10 = __builtin_amdgcn_wmma_f32_16x16x32_bf16(false, fa1h.v, false, fb0h.v, (short)0, acc10, false, false);
      acc10 = __builtin_amdgcn_wmma_f32_16x16x32_bf16(false, fa1h.v, false, fb0l.v, (short)0, acc10, false, false);
      acc10 = __builtin_amdgcn_wmma_f32_16x16x32_bf16(false, fa1l.v, false, fb0h.v, (short)0, acc10, false, false);
      acc11 = __builtin_amdgcn_wmma_f32_16x16x32_bf16(false, fa1h.v, false, fb1h.v, (short)0, acc11, false, false);
      acc11 = __builtin_amdgcn_wmma_f32_16x16x32_bf16(false, fa1h.v, false, fb1l.v, (short)0, acc11, false, false);
      acc11 = __builtin_amdgcn_wmma_f32_16x16x32_bf16(false, fa1l.v, false, fb1h.v, (short)0, acc11, false, false);
    }

    // fold into running argmin: dist = ||c||^2 - 2 x.c  (||x||^2 dropped).
    // padded cols: cnorm=+inf, acc=0 -> can never win; no predicates needed.
    float cn0 = (kcA < K) ? cnorm[kcA] : __builtin_inff();
    float cn1 = (kcB < K) ? cnorm[kcB] : __builtin_inff();
#pragma unroll
    for (int i = 0; i < 8; ++i) {
      float dA0 = cn0 - 2.f * acc00[i];
      float dB0 = cn1 - 2.f * acc01[i];
      float dv0 = dA0; int ki0 = kcA;
      if (dB0 < dA0) { dv0 = dB0; ki0 = kcB; }          // strict: lower k wins ties
      if (dv0 < bestD[i]) { bestD[i] = dv0; bestI[i] = ki0; }

      float dA1 = cn0 - 2.f * acc10[i];
      float dB1 = cn1 - 2.f * acc11[i];
      float dv1 = dA1; int ki1 = kcA;
      if (dB1 < dA1) { dv1 = dB1; ki1 = kcB; }
      if (dv1 < bestD[8 + i]) { bestD[8 + i] = dv1; bestI[8 + i] = ki1; }
    }
  }

  // reduce over the 16 column lanes (prefer lower index on ties)
#pragma unroll
  for (int mask = 1; mask < 16; mask <<= 1) {
#pragma unroll
    for (int i = 0; i < 16; ++i) {
      float od = __shfl_xor(bestD[i], mask, 32);
      int   oi = __shfl_xor(bestI[i], mask, 32);
      if (od < bestD[i] || (od == bestD[i] && oi < bestI[i])) { bestD[i] = od; bestI[i] = oi; }
    }
  }
  if ((lane & 15) == 0) {   // lanes 0 (rows +0..7) and 16 (rows +8..15)
#pragma unroll
    for (int i = 0; i < 8; ++i) {
      int r0 = rw * 32 + h * 8 + i;
      int r1 = r0 + 16;
      redD[cw][r0] = bestD[i];     redI[cw][r0] = bestI[i];
      redD[cw][r1] = bestD[8 + i]; redI[cw][r1] = bestI[8 + i];
    }
  }
  __syncthreads();
  if (t < TILE_ROWS) {
    float da = redD[0][t]; int ia = redI[0][t];
#pragma unroll
    for (int c = 1; c < 4; ++c) {
      float db = redD[c][t]; int ib = redI[c][t];
      if (db < da || (db == da && ib < ia)) { da = db; ia = ib; }
    }
    out[rowBase + t] = ia;
  }
}

extern "C" void kernel_launch(void* const* d_in, const int* in_sizes, int n_in,
                              void* d_out, int out_size, void* d_ws, size_t ws_size,
                              hipStream_t stream) {
  const float* x    = (const float*)d_in[0];
  const float* cent = (const float*)d_in[1];
  const int D = D_DIM;
  const int K = 10000;
  const int N = in_sizes[0] / D;
  const int Kpad = ((K + KTILE - 1) / KTILE) * KTILE;   // 10112

  // ws layout: [cnorm: K floats][chi: Kpad*D bf16][clo: Kpad*D bf16]  ~31 MB
  char* ws = (char*)d_ws;
  float* cnorm = (float*)ws;
  size_t off = ((size_t)K * sizeof(float) + 255) & ~(size_t)255;
  unsigned short* chi = (unsigned short*)(ws + off);
  unsigned short* clo = chi + (size_t)Kpad * D;

  hipLaunchKernelGGL(cnorm_kernel, dim3((K + 255) / 256), dim3(256), 0, stream,
                     cent, cnorm, K, D);
  hipLaunchKernelGGL(convert_centroids_kernel, dim3(Kpad / 64, D / 64), dim3(256), 0, stream,
                     cent, chi, clo, K);
  hipLaunchKernelGGL(kmeans_argmin_kernel, dim3(N / TILE_ROWS), dim3(256), 0, stream,
                     x, chi, clo, cnorm, (int*)d_out, K, Kpad);
}
